// RNN_49134425866917
// MI455X (gfx1250) — compile-verified, matrix operand order checked by reading, not run on hardware
//
#include <hip/hip_runtime.h>
#include <hip/hip_bf16.h>

// ---------------- Types for WMMA ----------------
typedef __bf16 bf16_t;
typedef bf16_t v16bf __attribute__((ext_vector_type(16)));
typedef bf16_t v8bf  __attribute__((ext_vector_type(8)));
typedef float  v8f   __attribute__((ext_vector_type(8)));

// ---------------- Problem constants ----------------
#define TT 256
#define BB 64
#define LL 4
#define HH 1024
#define INN 1024
#define KDIM 2048   // IN + H
#define NDIM 2048   // 2*H
#define BH (BB * HH)            // 65536 elements per [B,H] slab
#define WELEMS (KDIM * NDIM)    // per-layer weight elements

// ---------------- fp32 -> bf16 (round to nearest even) ----------------
__device__ __forceinline__ unsigned short f32_to_bf16(float f) {
    union { float f; unsigned int u; } v; v.f = f;
    unsigned int u = v.u;
    unsigned int r = u + 0x7FFFu + ((u >> 16) & 1u);
    return (unsigned short)(r >> 16);
}

// ---------------- fragment load: two contiguous b128 loads ----------------
__device__ __forceinline__ v16bf load_frag(const unsigned short* __restrict__ p) {
    v8bf lo = *(const v8bf*)(p);
    v8bf hi = *(const v8bf*)(p + 16);
    return __builtin_shufflevector(lo, hi,
        0, 1, 2, 3, 4, 5, 6, 7, 8, 9, 10, 11, 12, 13, 14, 15);
}

// ---------------- Prep: W [L,K,N] f32 -> Wt [L,N,K] bf16 (transpose) ----------------
__global__ void prep_weights(const float* __restrict__ W, unsigned short* __restrict__ Wt) {
    int idx = blockIdx.x * blockDim.x + threadIdx.x;   // over L*K*N
    int l = idx >> 22;                 // / (2048*2048)
    int rem = idx & (WELEMS - 1);
    int n = rem >> 11;                 // dest row (N)
    int k = rem & 2047;                // dest col (K)
    Wt[idx] = f32_to_bf16(W[l * WELEMS + k * NDIM + n]);
}

// ---------------- Prep: x f32 -> bf16 ----------------
__global__ void prep_x(const float* __restrict__ x, unsigned short* __restrict__ xb) {
    int idx = blockIdx.x * blockDim.x + threadIdx.x;
    xb[idx] = f32_to_bf16(x[idx]);
}

// ---------------- Prep: h init (f32 state + bf16 operand copy, parity 0) ----------------
__global__ void prep_h(const float* __restrict__ h_in, float* __restrict__ h_f32,
                       unsigned short* __restrict__ h_bf) {
    int idx = blockIdx.x * blockDim.x + threadIdx.x;
    float v = h_in[idx];
    h_f32[idx] = v;
    h_bf[idx] = f32_to_bf16(v);
}

// ---------------- Software-pipelined 1024-wide K panel ----------------
// ap  : A row base for this lane, already offset by half*8 (k=0 of panel)
// bp* : Wt row bases for the 4 N tiles, already offset by half*8 + panel K offset
__device__ __forceinline__ void gemm_panel(
    const unsigned short* __restrict__ ap,
    const unsigned short* __restrict__ bp0,
    const unsigned short* __restrict__ bp1,
    const unsigned short* __restrict__ bp2,
    const unsigned short* __restrict__ bp3,
    v8f (&acc)[4])
{
    v16bf a  = load_frag(ap);
    v16bf b0 = load_frag(bp0);
    v16bf b1 = load_frag(bp1);
    v16bf b2 = load_frag(bp2);
    v16bf b3 = load_frag(bp3);
    for (int k = 32; k < 1024; k += 32) {
        // prefetch next slab before consuming current one -> loads overlap WMMAs
        v16bf an  = load_frag(ap  + k);
        v16bf c0  = load_frag(bp0 + k);
        v16bf c1  = load_frag(bp1 + k);
        v16bf c2  = load_frag(bp2 + k);
        v16bf c3  = load_frag(bp3 + k);
        acc[0] = __builtin_amdgcn_wmma_f32_16x16x32_bf16(false, a, false, b0, (short)0, acc[0], false, false);
        acc[1] = __builtin_amdgcn_wmma_f32_16x16x32_bf16(false, a, false, b1, (short)0, acc[1], false, false);
        acc[2] = __builtin_amdgcn_wmma_f32_16x16x32_bf16(false, a, false, b2, (short)0, acc[2], false, false);
        acc[3] = __builtin_amdgcn_wmma_f32_16x16x32_bf16(false, a, false, b3, (short)0, acc[3], false, false);
        a = an; b0 = c0; b1 = c1; b2 = c2; b3 = c3;
    }
    acc[0] = __builtin_amdgcn_wmma_f32_16x16x32_bf16(false, a, false, b0, (short)0, acc[0], false, false);
    acc[1] = __builtin_amdgcn_wmma_f32_16x16x32_bf16(false, a, false, b1, (short)0, acc[1], false, false);
    acc[2] = __builtin_amdgcn_wmma_f32_16x16x32_bf16(false, a, false, b2, (short)0, acc[2], false, false);
    acc[3] = __builtin_amdgcn_wmma_f32_16x16x32_bf16(false, a, false, b3, (short)0, acc[3], false, false);
}

// ---------------- Fused GEMM + SRU gate ----------------
// z = [inpA | hA] @ W + b  (computed via WMMA, bf16 inputs / f32 accum)
// hn = sigmoid(z_f) * h_prev + (1 - sigmoid(z_f)) * tanh(z_l)
// Each wave owns a 16(M) x 32(N-learn) strip PLUS the matching forget strip at
// N+1024, so learn/forget pairs land in the same lane & accumulator slot.
// Grid: 16 blocks x 256 threads (8 waves) = 128 waves.
__global__ void __launch_bounds__(256)
sru_step(const unsigned short* __restrict__ inpA,   // [64,1024] bf16 (K = 0..1023)
         const unsigned short* __restrict__ hA,     // [64,1024] bf16 (K = 1024..2047), parity RP
         const unsigned short* __restrict__ Wt,     // [2048,2048] bf16 (pre-transposed [N][K])
         const float* __restrict__ bias,            // [2048]
         const float* __restrict__ h_rd,            // [64,1024] f32 state, parity RP
         float* __restrict__ h_wr,                  // [64,1024] f32 state, parity WP
         unsigned short* __restrict__ hb_wr,        // [64,1024] bf16 operand, parity WP
         unsigned short* __restrict__ next_inp,     // [64,1024] bf16 or nullptr
         float* __restrict__ out_t)                 // [64,1024] f32 or nullptr
{
    const int wave = threadIdx.x >> 5;
    const int lane = threadIdx.x & 31;
    const int tile = blockIdx.x * 8 + wave;        // 0..127
    const int m_tile = tile & 3;                   // 4 row tiles of 16
    const int n0 = (tile >> 2) * 32;               // 32 learn strips of 32 cols
    const int half = lane >> 4;
    const int lr = lane & 15;
    const int m = m_tile * 16 + lr;

    v8f acc[4];                                    // [0,1]=learn n0,n0+16  [2,3]=forget +1024
    acc[0] = v8f{}; acc[1] = v8f{}; acc[2] = v8f{}; acc[3] = v8f{};

    const int ko = half * 8;                       // per-lane K sub-offset (ISA bf16 layout)
    const unsigned short* w0 = Wt + (size_t)(n0 + lr)        * KDIM + ko;
    const unsigned short* w1 = Wt + (size_t)(n0 + 16 + lr)   * KDIM + ko;
    const unsigned short* w2 = Wt + (size_t)(n0 + 1024 + lr) * KDIM + ko;
    const unsigned short* w3 = Wt + (size_t)(n0 + 1040 + lr) * KDIM + ko;

    // Panel 1: K = 0..1023  (layer input)
    gemm_panel(inpA + m * INN + ko, w0, w1, w2, w3, acc);
    // Panel 2: K = 1024..2047 (recurrent h)
    gemm_panel(hA + m * HH + ko, w0 + 1024, w1 + 1024, w2 + 1024, w3 + 1024, acc);

    // Fused gate epilogue. C/D layout: VGPR v -> M = m_tile*16 + v + half*8 ; lane lr -> N
#pragma unroll
    for (int j = 0; j < 2; ++j) {
        const int n = n0 + j * 16 + lr;            // learn column (0..1023)
        const float bl = bias[n];
        const float bf_ = bias[n + 1024];
#pragma unroll
        for (int v = 0; v < 8; ++v) {
            const int mm = m_tile * 16 + v + half * 8;
            const int i = mm * HH + n;
            const float learn  = acc[j][v] + bl;
            const float forget = acc[j + 2][v] + bf_;
            const float f = 1.0f / (1.0f + __expf(-forget));
            const float hp = h_rd[i];
            const float hn = f * hp + (1.0f - f) * tanhf(learn);
            h_wr[i] = hn;
            const unsigned short hb = f32_to_bf16(hn);
            hb_wr[i] = hb;
            if (next_inp) next_inp[i] = hb;
            if (out_t)    out_t[i] = hn;
        }
    }
}

// ---------------- Copy h_final to the tail of d_out ----------------
__global__ void copy_tail(const float* __restrict__ src, float* __restrict__ dst) {
    int i = blockIdx.x * blockDim.x + threadIdx.x;
    dst[i] = src[i];
}

extern "C" void kernel_launch(void* const* d_in, const int* in_sizes, int n_in,
                              void* d_out, int out_size, void* d_ws, size_t ws_size,
                              hipStream_t stream) {
    const float* x  = (const float*)d_in[0];   // [T,B,IN]
    const float* h0 = (const float*)d_in[1];   // [L,B,H]
    const float* W  = (const float*)d_in[2];   // [L,K,N]
    const float* b  = (const float*)d_in[3];   // [L,N]
    float* out = (float*)d_out;                // outputs [T,B,H] ++ h_final [L*B,H]

    // ---- workspace carve-up (all offsets large-power-of-2 aligned) ----
    char* ws = (char*)d_ws;
    size_t off = 0;
    unsigned short* Wt      = (unsigned short*)(ws + off); off += (size_t)LL * WELEMS * 2;     // 32 MB
    unsigned short* xb      = (unsigned short*)(ws + off); off += (size_t)TT * BH * 2;         // 32 MB
    float*          h32a    = (float*)(ws + off);          off += (size_t)LL * BH * 4;         // 1 MB (parity 0)
    float*          h32b    = (float*)(ws + off);          off += (size_t)LL * BH * 4;         // 1 MB (parity 1)
    unsigned short* hbfa    = (unsigned short*)(ws + off); off += (size_t)LL * BH * 2;         // 0.5 MB
    unsigned short* hbfb    = (unsigned short*)(ws + off); off += (size_t)LL * BH * 2;         // 0.5 MB
    unsigned short* inpb    = (unsigned short*)(ws + off); off += (size_t)LL * BH * 2;         // 0.5 MB (layers 1..3)

    float*          h32p[2] = { h32a, h32b };
    unsigned short* hbfp[2] = { hbfa, hbfb };

    // ---- one-time prep (inside graph; deterministic) ----
    prep_weights<<<(LL * WELEMS) / 256, 256, 0, stream>>>(W, Wt);
    prep_x<<<(TT * BH) / 256, 256, 0, stream>>>(x, xb);
    prep_h<<<(LL * BH) / 256, 256, 0, stream>>>(h0, h32a, hbfa);   // parity 0

    // ---- sequential recurrence: 256 steps x 4 layers, ping-pong h buffers ----
    for (int t = 0; t < TT; ++t) {
        const int rp = t & 1;          // read parity
        const int wp = rp ^ 1;         // write parity
        for (int l = 0; l < LL; ++l) {
            const unsigned short* inpA =
                (l == 0) ? (xb + (size_t)t * BH) : (inpb + (size_t)l * BH);
            unsigned short* nxt = (l < LL - 1) ? (inpb + (size_t)(l + 1) * BH) : nullptr;
            float* out_t = (l == LL - 1) ? (out + (size_t)t * BH) : nullptr;
            sru_step<<<16, 256, 0, stream>>>(
                inpA,
                hbfp[rp] + (size_t)l * BH,
                Wt + (size_t)l * WELEMS,
                b + (size_t)l * NDIM,
                h32p[rp] + (size_t)l * BH,
                h32p[wp] + (size_t)l * BH,
                hbfp[wp] + (size_t)l * BH,
                nxt, out_t);
        }
    }

    // T=256 is even -> final state sits in parity 0 buffers
    copy_tail<<<(LL * BH) / 256, 256, 0, stream>>>(h32a, out + (size_t)TT * BH);
}
